// ConvBlock_64192581206728
// MI455X (gfx1250) — matrix operand order repaired and checked by reading, not compiled, not run on hardware
//
#include <hip/hip_runtime.h>
#include <hip/hip_bf16.h>

typedef __attribute__((ext_vector_type(16))) _Float16 v16h;
typedef __attribute__((ext_vector_type(8)))  float    v8f;

#define SP 524288            // 8*16*64*64 spatial positions
#define NEG_SLOPE 0.2f
#define INV_SP (1.0f/524288.0f)

// ---------------------------------------------------------------------------
// Pack conv weights (Cout=32, Cin, 3,3,3,3) into WMMA A-fragment lane layout.
// A 16x32 f16 layout (contiguous K, matches B): lanes 0-15: halves j -> K=j,
// lanes 16-31: K = 16+j.  M = lane%16.
// K within a chunk = tapInChunk*CIN + cin; chunk cc covers taps
// [cc*TPC, cc*TPC+TPC) local to the current da-slice (27 taps per slice).
// apack flat layout: [da*NCDA+cc][tile(2)][lane(32)][j(16)] halves.
// ---------------------------------------------------------------------------
template<int CIN, int TPC, int NCDA>
__global__ void pack_weights(const float* __restrict__ w,
                             _Float16* __restrict__ apack) {
  int total = 3 * NCDA * 2 * 32 * 16;
  for (int i = blockIdx.x * blockDim.x + threadIdx.x; i < total;
       i += gridDim.x * blockDim.x) {
    int j     = i & 15;
    int lane  = (i >> 4) & 31;
    int tile  = (i >> 9) & 1;
    int chunk = i >> 10;
    int da = chunk / NCDA, cc = chunk % NCDA;
    int m   = lane & 15;
    int klo = j;
    int K   = klo + ((lane >= 16) ? 16 : 0);
    int tic = K / CIN, cin = K % CIN;
    int lt  = cc * TPC + tic;          // local tap within da slice (27 taps)
    float v = 0.0f;
    if (lt < 27) {
      int tap  = da * 27 + lt;         // tap = da*27 + db*9 + dh*3 + dw
      int cout = tile * 16 + m;
      v = w[(size_t)(cout * CIN + cin) * 81 + tap];
    }
    apack[i] = (_Float16)v;
  }
}

// ---------------------------------------------------------------------------
// Implicit-GEMM 4D conv via v_wmma_f32_16x16x32_f16.
// One block per output line (a,b,h); 4 waves, each wave: 16 W positions x
// 32 Cout (two 16-row WMMA tiles).  Stages 9 input rows (db,dh) x 66 cols x
// CIN channels per da-slice into LDS as f16 (optionally normalizing +
// leaky-ReLU the layer-1 output on the fly for conv2).
// B fragment (32x16 K-major): lanes 0-15 hold K=0..15, lanes 16-31 K=16..31,
// N = lane%16 -> one contiguous 32B LDS read per lane per chunk.
// ---------------------------------------------------------------------------
template<int CIN, int TPC, int NCDA, bool NORM>
__global__ void __launch_bounds__(128)
conv4d_wmma(const float* __restrict__ x,
            const _Float16* __restrict__ apack,
            const float* __restrict__ mean,
            const float* __restrict__ scale,
            float* __restrict__ y) {
  __shared__ __align__(32) _Float16 xs[9 * 66 * CIN];

  int tid  = threadIdx.x;
  int a    = blockIdx.x >> 10;
  int b    = (blockIdx.x >> 6) & 15;
  int h    = blockIdx.x & 63;
  int lane = tid & 31;
  int wave = tid >> 5;
  int wbase = wave << 4;       // W segment base for this wave
  int n  = lane & 15;          // N column within tile
  int hi = lane >> 4;          // lane half

  v8f c0 = {};                 // Cout 0..15 accumulator
  v8f c1 = {};                 // Cout 16..31 accumulator

  for (int da = 0; da < 3; ++da) {
    if (da) __syncthreads();
    // ---- stage da-slice into LDS (fp32 -> fp16, + norm/leaky for layer 2)
    for (int s = tid; s < 9 * CIN * 66; s += 128) {
      int col  = s % 66;                 // input w + 1
      int rest = s / 66;
      int cin  = rest % CIN;
      int rl   = rest / CIN;             // rl = db*3 + dh
      int ia = a + da - 1;
      int ib = b + rl / 3 - 1;
      int ih = h + rl % 3 - 1;
      int iw = col - 1;
      float v = 0.0f;
      if ((unsigned)ia < 8u && (unsigned)ib < 16u &&
          (unsigned)ih < 64u && (unsigned)iw < 64u) {
        v = x[(size_t)cin * SP + ia * 65536 + ib * 4096 + ih * 64 + iw];
        if constexpr (NORM) {
          v = (v - mean[cin]) * scale[cin];
          v = (v >= 0.0f) ? v : NEG_SLOPE * v;
        }
      }
      xs[(rl * 66 + col) * CIN + cin] = (_Float16)v;
    }
    __syncthreads();

    // ---- K loop over this da-slice's chunks
    const _Float16* ap = apack + (size_t)(da * NCDA) * 1024 + lane * 16;
    for (int cc = 0; cc < NCDA; ++cc) {
      v16h a0 = *(const v16h*)(ap);          // tile 0 A fragment (32B)
      v16h a1 = *(const v16h*)(ap + 512);    // tile 1 A fragment
      __builtin_prefetch(ap + 1024, 0, 0);   // next chunk's A stream
      v16h bf = {};
      int lt = cc * TPC + ((TPC == 2) ? hi : 0);   // per-lane tap
      if (lt < 27) {
        int rl  = lt / 3, dw = lt % 3;
        int col = wbase + n + dw;
        int cinb = (TPC == 1) ? (hi << 4) : 0;
        bf = *(const v16h*)(&xs[(rl * 66 + col) * CIN + cinb]);
      }
      c0 = __builtin_amdgcn_wmma_f32_16x16x32_f16(false, a0, false, bf,
                                                  (short)0, c0, false, false);
      c1 = __builtin_amdgcn_wmma_f32_16x16x32_f16(false, a1, false, bf,
                                                  (short)0, c1, false, false);
      ap += 1024;
    }
  }

  // ---- store raw conv output (C/D layout: VGPR r -> M = r + 8*hi, N = n)
  int posb = a * 65536 + b * 4096 + h * 64 + wbase + n;
#pragma unroll
  for (int r = 0; r < 8; ++r) {
    int m = r + (hi << 3);
    y[(size_t)m * SP + posb]        = c0[r];
    y[(size_t)(16 + m) * SP + posb] = c1[r];
  }
}

// ---------------------------------------------------------------------------
// Deterministic per-channel mean / rsqrt(var+eps): one block per channel,
// fixed-order strided accumulation + shared-memory tree reduction.
// ---------------------------------------------------------------------------
__global__ void instnorm_stats(const float* __restrict__ y,
                               float* __restrict__ mean,
                               float* __restrict__ scale) {
  __shared__ float ss[256], sq[256];
  int c = blockIdx.x, t = threadIdx.x;
  const float* p = y + (size_t)c * SP;
  float s = 0.0f, q = 0.0f;
  for (int i = t; i < SP; i += 256) {
    float v = p[i];
    s += v;
    q += v * v;
  }
  ss[t] = s; sq[t] = q;
  __syncthreads();
  for (int o = 128; o > 0; o >>= 1) {
    if (t < o) { ss[t] += ss[t + o]; sq[t] += sq[t + o]; }
    __syncthreads();
  }
  if (t == 0) {
    float m = ss[0] * INV_SP;
    float v = sq[0] * INV_SP - m * m;
    mean[c]  = m;
    scale[c] = rsqrtf(v + 1e-5f);
  }
}

// ---------------------------------------------------------------------------
// Final elementwise norm + leaky-ReLU (in place on d_out).
// ---------------------------------------------------------------------------
__global__ void norm_leaky(const float* __restrict__ yin,
                           const float* __restrict__ mean,
                           const float* __restrict__ scale,
                           float* __restrict__ out, int total) {
  int i = blockIdx.x * blockDim.x + threadIdx.x;
  if (i < total) {
    int c = i >> 19;                       // / SP
    float v = (yin[i] - mean[c]) * scale[c];
    out[i] = (v >= 0.0f) ? v : NEG_SLOPE * v;
  }
}

// ---------------------------------------------------------------------------
extern "C" void kernel_launch(void* const* d_in, const int* in_sizes, int n_in,
                              void* d_out, int out_size, void* d_ws,
                              size_t ws_size, hipStream_t stream) {
  const float* image = (const float*)d_in[0];   // (1,16,8,16,64,64)
  const float* w1    = (const float*)d_in[1];   // (32,16,3,3,3,3)
  const float* w2    = (const float*)d_in[2];   // (32,32,3,3,3,3)
  float* out = (float*)d_out;                   // (1,32,8,16,64,64)

  char* ws = (char*)d_ws;
  float* y1 = (float*)ws;                                    // 64 MB raw conv1
  size_t off = (size_t)32 * SP * sizeof(float);
  _Float16* apack1 = (_Float16*)(ws + off);                  // 42*1024 halves
  off += (size_t)42 * 1024 * sizeof(_Float16);
  off = (off + 255) & ~(size_t)255;
  _Float16* apack2 = (_Float16*)(ws + off);                  // 81*1024 halves
  off += (size_t)81 * 1024 * sizeof(_Float16);
  off = (off + 255) & ~(size_t)255;
  float* stats  = (float*)(ws + off);
  float* mean1  = stats;       float* scale1 = stats + 32;
  float* mean2  = stats + 64;  float* scale2 = stats + 96;

  // 1) pack weights into WMMA A-fragment layout (f16)
  pack_weights<16, 2, 14><<<168, 256, 0, stream>>>(w1, apack1);
  pack_weights<32, 1, 27><<<324, 256, 0, stream>>>(w2, apack2);

  // 2) conv1: fp32 image -> raw fp32 conv output (f16 WMMA, f32 accumulate)
  conv4d_wmma<16, 2, 14, false><<<8192, 128, 0, stream>>>(
      image, apack1, nullptr, nullptr, y1);

  // 3) instance-norm stats for layer 1
  instnorm_stats<<<32, 256, 0, stream>>>(y1, mean1, scale1);

  // 4) conv2: normalize+leaky y1 on the fly while staging; raw out -> d_out
  conv4d_wmma<32, 1, 27, true><<<8192, 128, 0, stream>>>(
      y1, apack2, mean1, scale1, out);

  // 5) instance-norm stats for layer 2
  instnorm_stats<<<32, 256, 0, stream>>>(out, mean2, scale2);

  // 6) final norm + leaky in place
  norm_leaky<<<(32 * SP + 255) / 256, 256, 0, stream>>>(out, mean2, scale2,
                                                        out, 32 * SP);
}
